// GNNEncoder_9715216023654
// MI455X (gfx1250) — matrix operand order repaired and checked by reading, not compiled, not run on hardware
//
#include <hip/hip_runtime.h>
#include <stdint.h>

#define NN   65536
#define NE   1048576
#define FN   133
#define FE   14
#define HID  128
#define NG   256

// ---------- types ----------
typedef __attribute__((ext_vector_type(16))) __bf16        v16bf;
typedef __attribute__((ext_vector_type(8)))  float         v8f;
typedef __attribute__((ext_vector_type(4)))  float         v4f;
typedef __attribute__((ext_vector_type(4)))  unsigned int  v4u;
typedef __attribute__((ext_vector_type(4)))  int           v4i;

union AFrag { v16bf bf; unsigned short us[16]; unsigned u32[8]; v4u q[2]; };
union US8   { v4u q; unsigned short us[8]; unsigned u32[4]; };

__device__ __forceinline__ unsigned short f2bf(float f) {   // RNE (weight pack only)
  unsigned u = __builtin_bit_cast(unsigned, f);
  u += 0x7FFFu + ((u >> 16) & 1u);
  return (unsigned short)(u >> 16);
}
__device__ __forceinline__ unsigned short f2bf_rtz(float f) {   // 1-VALU truncating convert
  return (unsigned short)(__builtin_bit_cast(unsigned, f) >> 16);
}
__device__ __forceinline__ float bf2f(unsigned short s) {
  unsigned u = ((unsigned)s) << 16;
  return __builtin_bit_cast(float, u);
}
// pack two fp32 -> bf16x2 dword (a -> low half, b -> high half)
#if __has_builtin(__builtin_amdgcn_cvt_pk_bf16_f32)
__device__ __forceinline__ unsigned pk_bf(float a, float b) {   // hw packed convert (RNE)
  auto r = __builtin_amdgcn_cvt_pk_bf16_f32(a, b);
  return __builtin_bit_cast(unsigned, r);
}
#elif __has_builtin(__builtin_amdgcn_perm)
__device__ __forceinline__ unsigned pk_bf(float a, float b) {   // single v_perm_b32 (RTZ)
  return __builtin_amdgcn_perm(__builtin_bit_cast(unsigned, b),
                               __builtin_bit_cast(unsigned, a), 0x07060302u);
}
#else
__device__ __forceinline__ unsigned pk_bf(float a, float b) {
  return (unsigned)(__builtin_bit_cast(unsigned, a) >> 16) |
         (__builtin_bit_cast(unsigned, b) & 0xFFFF0000u);
}
#endif
__device__ __forceinline__ v8f vzero() {
  v8f z = {0.f,0.f,0.f,0.f,0.f,0.f,0.f,0.f}; return z;
}
__device__ __forceinline__ v16bf load_frag(const unsigned short* p) {
  AFrag f; const v4u* q = (const v4u*)p; f.q[0] = q[0]; f.q[1] = q[1]; return f.bf;
}
#define WMMA_BF16(a, b, c) \
  __builtin_amdgcn_wmma_f32_16x16x32_bf16(false, (a), false, (b), (short)0, (c), false, false)

#if __has_builtin(__builtin_amdgcn_global_load_async_to_lds_b128)
#define ASYNC_LDS 1
typedef __attribute__((address_space(1))) v4i gv4i;   // global (prints as __device__)
typedef __attribute__((address_space(3))) v4i lv4i;   // LDS
#else
#define ASYNC_LDS 0
#endif

// ---------- workspace layout (bytes) ----------
#define OFF_H    ((size_t)0)                        // bf16 [NE][128]   256 MB
#define OFF_H0   ((size_t)268435456)                // bf16 [NE][128]   256 MB
#define OFF_AGG  ((size_t)536870912)                // f32  [NN][128]    32 MB
#define OFF_XWE  ((size_t)570425344)                // f32  [NN][128]    32 MB
#define OFF_XWN  ((size_t)603979776)                // f32  [NN][128]    32 MB
#define OFF_WP   ((size_t)637534208)                // packed bf16 weights
// packed-weight sub-offsets (ushort units)
#define WXP   0        // x-part of We,  K=160 (5 ktiles)
#define WNXP  20480    // x-part of Wn,  K=160 (5 ktiles)
#define WEEP  40960    // edge part We,  K=32  (1 ktile)
#define WCP   45056    // Wc[l],         K=128 (4 ktiles), 16384 each
#define WNSP  94208    // s-part of Wn,  K=128 (4 ktiles)

// Pack fp32 weights [K x 128] into WMMA B-fragment order:
// idx = ((kt*8+nt)*32 + lane)*16 + j  ->  W[row0 + kt*32 + (j<8 ? (lane>>4)*8 + j : 16 + (lane>>4)*8 + j-8)][nt*16 + (lane&15)]
__global__ void k_pack(const float* __restrict__ We, const float* __restrict__ Wc,
                       const float* __restrict__ Wn, unsigned short* __restrict__ wp) {
  int job = blockIdx.x;
  const float* src; int row0, nrows, ktc, dst;
  switch (job) {
    case 0:  src = We; row0 = 0;   nrows = 133; ktc = 5; dst = WXP;  break;
    case 1:  src = Wn; row0 = 0;   nrows = 133; ktc = 5; dst = WNXP; break;
    case 2:  src = We; row0 = 133; nrows = 14;  ktc = 1; dst = WEEP; break;
    case 3: case 4: case 5:
             src = Wc + (size_t)(job-3)*HID*HID; row0 = 0; nrows = 128; ktc = 4;
             dst = WCP + (job-3)*16384; break;
    default: src = Wn; row0 = 133; nrows = 128; ktc = 4; dst = WNSP; break;
  }
  int total = ktc * 4096;
  for (int i = threadIdx.x; i < total; i += blockDim.x) {
    int kt = i >> 12, rem = i & 4095;
    int nt = rem >> 9; rem &= 511;
    int lane = rem >> 4, j = rem & 15;
    int col = nt*16 + (lane & 15);
    int k   = kt*32 + (j < 8 ? (lane >> 4)*8 + j : 16 + (lane >> 4)*8 + (j - 8));
    float v = (k < nrows) ? src[(size_t)(row0 + k)*HID + col] : 0.f;
    wp[dst + i] = f2bf(v);
  }
}

__global__ void k_zero4(float* __restrict__ p, int n4) {
  int i = blockIdx.x*blockDim.x + threadIdx.x;
  v4f z = {0.f,0.f,0.f,0.f};
  if (i < n4) ((v4f*)p)[i] = z;
}

// issue 4 B-frag loads then 4 WMMAs (caps register liveness)
#define WMMA_GROUP4(bk, g, a, c)                                     \
  {                                                                  \
    v16bf b0 = load_frag((bk) + ((g)*4 + 0)*512);                    \
    v16bf b1 = load_frag((bk) + ((g)*4 + 1)*512);                    \
    v16bf b2 = load_frag((bk) + ((g)*4 + 2)*512);                    \
    v16bf b3 = load_frag((bk) + ((g)*4 + 3)*512);                    \
    (c)[(g)*4 + 0] = WMMA_BF16((a).bf, b0, (c)[(g)*4 + 0]);          \
    (c)[(g)*4 + 1] = WMMA_BF16((a).bf, b1, (c)[(g)*4 + 1]);          \
    (c)[(g)*4 + 2] = WMMA_BF16((a).bf, b2, (c)[(g)*4 + 2]);          \
    (c)[(g)*4 + 3] = WMMA_BF16((a).bf, b3, (c)[(g)*4 + 3]);          \
  }

// C[NN x 128] = x[NN x 133] @ Bpack(K padded to 160)
__global__ void k_node_gemm133(const float* __restrict__ A,
                               const unsigned short* __restrict__ Bp,
                               float* __restrict__ C) {
  int wave = threadIdx.x >> 5, lane = threadIdx.x & 31;
  int tile = blockIdx.x*(blockDim.x >> 5) + wave;     // 16-row tile
  int m = lane & 15, half = lane >> 4;
  const float* arow = A + (size_t)(tile*16 + m)*FN;
  const unsigned short* bp = Bp + (size_t)lane*16;
  v8f c[8];
  #pragma unroll
  for (int nt = 0; nt < 8; ++nt) c[nt] = vzero();
  #pragma unroll 1
  for (int kt = 0; kt < 4; ++kt) {                    // k < 128 < 133: unmasked
    int kb = kt*32 + half*8;
    AFrag a;
    #pragma unroll
    for (int j = 0; j < 4; ++j) {                     // scalar loads: 4B-aligned rows
      a.u32[j]     = pk_bf(arow[kb + 2*j],      arow[kb + 2*j + 1]);
      a.u32[4 + j] = pk_bf(arow[kb + 16 + 2*j], arow[kb + 17 + 2*j]);
    }
    const unsigned short* bk = bp + (size_t)kt*4096;
    WMMA_GROUP4(bk, 0, a, c);
    WMMA_GROUP4(bk, 1, a, c);
  }
  {                                                   // kt = 4: masked tail, no branches
    int kb = 128 + half*8;
    float t8[8];
    #pragma unroll
    for (int j = 0; j < 8; ++j) {
      int k  = kb + j;
      int kc = (k < FN) ? k : 0;                      // clamp index, select value
      float v = arow[kc];
      t8[j] = (k < FN) ? v : 0.f;
    }
    AFrag a;
    #pragma unroll
    for (int j = 0; j < 4; ++j) { a.u32[j] = pk_bf(t8[2*j], t8[2*j+1]); a.u32[4+j] = 0u; }
    const unsigned short* bk = bp + (size_t)4*4096;
    WMMA_GROUP4(bk, 0, a, c);
    WMMA_GROUP4(bk, 1, a, c);
  }
  #pragma unroll
  for (int nt = 0; nt < 8; ++nt) {
    int col = nt*16 + m;
    #pragma unroll
    for (int i = 0; i < 8; ++i)
      C[(size_t)(tile*16 + half*8 + i)*HID + col] = c[nt][i];
  }
}

// h0[e] = h[e] = relu( xWe[row[e]] + edge_attr[e] @ We_edge + be )
__global__ void k_edge_init(const float* __restrict__ ea, const int* __restrict__ erow,
                            const float* __restrict__ xWe, const float* __restrict__ be,
                            const unsigned short* __restrict__ Bp,
                            unsigned short* __restrict__ h, unsigned short* __restrict__ h0) {
  __shared__ __align__(16) unsigned short lB[4096];       //  8 KB packed We_edge
  __shared__ __align__(16) unsigned short lT[8][2048];    // 32 KB transpose staging
#if ASYNC_LDS
  {
    gv4i* g = (gv4i*)Bp;                                  // async DMA: global -> LDS
    lv4i* l = (lv4i*)lB;
    for (int i = threadIdx.x; i < 512; i += blockDim.x)
      __builtin_amdgcn_global_load_async_to_lds_b128(g + i, l + i, 0, 0);
    asm volatile("s_wait_asynccnt 0" ::: "memory");
  }
#else
  {
    const v4u* s = (const v4u*)Bp; v4u* d = (v4u*)lB;
    for (int i = threadIdx.x; i < 512; i += blockDim.x) d[i] = s[i];
  }
#endif
  __syncthreads();

  int wave = threadIdx.x >> 5, lane = threadIdx.x & 31;
  int tile = blockIdx.x*(blockDim.x >> 5) + wave;         // 16-edge tile
  int m = lane & 15, half = lane >> 4;
  const float* er = ea + (size_t)(tile*16 + m)*FE;
  float t8[8];
  #pragma unroll
  for (int j = 0; j < 8; ++j) {                           // K=32 tile, 14 valid
    int k  = half*8 + j;
    int kc = (k < FE) ? k : 0;
    float v = er[kc];
    t8[j] = (k < FE) ? v : 0.f;
  }
  AFrag a;
  #pragma unroll
  for (int j = 0; j < 4; ++j) { a.u32[j] = pk_bf(t8[2*j], t8[2*j+1]); a.u32[4+j] = 0u; }
  v8f c[8];
  #pragma unroll
  for (int nt = 0; nt < 8; ++nt) c[nt] = vzero();
  const unsigned short* bk = lB + (size_t)lane*16;
  WMMA_GROUP4(bk, 0, a, c);
  WMMA_GROUP4(bk, 1, a, c);

  // wave-local LDS transpose -> row-chunk epilogue with b128 stores
  unsigned short* lw = &lT[wave][0];
  #pragma unroll
  for (int nt = 0; nt < 8; ++nt)
    #pragma unroll
    for (int i = 0; i < 8; ++i)
      lw[(half*8 + i)*HID + nt*16 + m] = f2bf_rtz(c[nt][i]);
  #pragma unroll
  for (int t = 0; t < 8; ++t) {
    int r = t*2 + half, cc = m*8;
    int e2 = tile*16 + r;
    int node = erow[e2];
    US8 cv; cv.q = *(const v4u*)(lw + r*HID + cc);
    const float* xr = xWe + (size_t)node*HID + cc;
    v4f x0 = *(const v4f*)xr,        x1 = *(const v4f*)(xr + 4);
    v4f b0 = *(const v4f*)(be + cc), b1 = *(const v4f*)(be + cc + 4);
    float vals[8];
    #pragma unroll
    for (int j = 0; j < 4; ++j) {
      vals[j]     = fmaxf(bf2f(cv.us[j])     + x0[j] + b0[j], 0.f);
      vals[4 + j] = fmaxf(bf2f(cv.us[4 + j]) + x1[j] + b1[j], 0.f);
    }
    US8 o;
    #pragma unroll
    for (int j = 0; j < 4; ++j) o.u32[j] = pk_bf(vals[2*j], vals[2*j+1]);
    *(v4u*)(h  + (size_t)e2*HID + cc) = o.q;
    *(v4u*)(h0 + (size_t)e2*HID + cc) = o.q;
  }
}

// agg[col[e]] += h[e]   (bf16 -> f32 atomics; agg is L2-resident)
__global__ void k_scatter(const unsigned short* __restrict__ h,
                          const int* __restrict__ ecol, float* __restrict__ agg) {
  size_t t = (size_t)blockIdx.x*blockDim.x + threadIdx.x;   // NE*16 threads
  int e = (int)(t >> 4), part = (int)(t & 15);
  US8 f; f.q = *(const v4u*)(h + (size_t)e*HID + part*8);
  int node = ecol[e];
  float* dst = agg + (size_t)node*HID + part*8;
  #pragma unroll
  for (int j = 0; j < 8; ++j) unsafeAtomicAdd(dst + j, bf2f(f.us[j]));
}

// h[e] = relu( (agg[row[e]] - h[e^1]) @ Wc[l] + bc[l] + h0[e] ), in place
__global__ void k_conv(const float* __restrict__ agg, const unsigned short* __restrict__ Bp,
                       const float* __restrict__ bcl, const int* __restrict__ erow,
                       unsigned short* __restrict__ h, const unsigned short* __restrict__ h0) {
  __shared__ __align__(16) unsigned short lT[8][2048];    // 32 KB transpose staging
  int wave = threadIdx.x >> 5, lane = threadIdx.x & 31;
  int tile = blockIdx.x*(blockDim.x >> 5) + wave;
  int m = lane & 15, half = lane >> 4;
  int e = tile*16 + m;
  const float*          ar = agg + (size_t)erow[e]*HID;   // 512B aligned
  const unsigned short* rv = h   + (size_t)(e ^ 1)*HID;   // 256B aligned
  const unsigned short* bp = Bp + (size_t)lane*16;
  __builtin_prefetch(h0 + (size_t)(tile*16)*HID, 0, 0);
  v8f c[8];
  #pragma unroll
  for (int nt = 0; nt < 8; ++nt) c[nt] = vzero();
  #pragma unroll 1
  for (int kt = 0; kt < 4; ++kt) {
    int kb = kt*32 + half*8;
    US8 r0, r1;
    r0.q = *(const v4u*)(rv + kb);
    r1.q = *(const v4u*)(rv + kb + 16);
    v4f A0 = *(const v4f*)(ar + kb),      A1 = *(const v4f*)(ar + kb + 4);
    v4f A2 = *(const v4f*)(ar + kb + 16), A3 = *(const v4f*)(ar + kb + 20);
    AFrag a;
    #pragma unroll
    for (int j = 0; j < 2; ++j) {
      a.u32[j]     = pk_bf(A0[2*j] - bf2f(r0.us[2*j]),     A0[2*j+1] - bf2f(r0.us[2*j+1]));
      a.u32[2 + j] = pk_bf(A1[2*j] - bf2f(r0.us[4 + 2*j]), A1[2*j+1] - bf2f(r0.us[5 + 2*j]));
      a.u32[4 + j] = pk_bf(A2[2*j] - bf2f(r1.us[2*j]),     A2[2*j+1] - bf2f(r1.us[2*j+1]));
      a.u32[6 + j] = pk_bf(A3[2*j] - bf2f(r1.us[4 + 2*j]), A3[2*j+1] - bf2f(r1.us[5 + 2*j]));
    }
    const unsigned short* bk = bp + (size_t)kt*4096;
    WMMA_GROUP4(bk, 0, a, c);
    WMMA_GROUP4(bk, 1, a, c);
  }
  // wave-local LDS transpose -> row-chunk epilogue (b128 h0 reads, b128 h writes)
  unsigned short* lw = &lT[wave][0];
  #pragma unroll
  for (int nt = 0; nt < 8; ++nt)
    #pragma unroll
    for (int i = 0; i < 8; ++i)
      lw[(half*8 + i)*HID + nt*16 + m] = f2bf_rtz(c[nt][i]);
  #pragma unroll
  for (int t = 0; t < 8; ++t) {
    int r = t*2 + half, cc = m*8;
    int e2 = tile*16 + r;
    US8 cv; cv.q = *(const v4u*)(lw + r*HID + cc);
    US8 hv; hv.q = *(const v4u*)(h0 + (size_t)e2*HID + cc);
    v4f b0 = *(const v4f*)(bcl + cc), b1 = *(const v4f*)(bcl + cc + 4);
    float vals[8];
    #pragma unroll
    for (int j = 0; j < 4; ++j) {
      vals[j]     = fmaxf(bf2f(cv.us[j])     + b0[j] + bf2f(hv.us[j]), 0.f);
      vals[4 + j] = fmaxf(bf2f(cv.us[4 + j]) + b1[j] + bf2f(hv.us[4 + j]), 0.f);
    }
    US8 o;
    #pragma unroll
    for (int j = 0; j < 4; ++j) o.u32[j] = pk_bf(vals[2*j], vals[2*j+1]);
    *(v4u*)(h + (size_t)e2*HID + cc) = o.q;
  }
}

// q = relu( xWn + s @ Wn_s + bn );  out[batch[n]] += q
__global__ void k_final(const float* __restrict__ s, const unsigned short* __restrict__ Bp,
                        const float* __restrict__ xWn, const float* __restrict__ bn,
                        const int* __restrict__ batch, float* __restrict__ out) {
  int wave = threadIdx.x >> 5, lane = threadIdx.x & 31;
  int tile = blockIdx.x*(blockDim.x >> 5) + wave;     // 16-node tile
  int m = lane & 15, half = lane >> 4;
  const float* ar = s + (size_t)(tile*16 + m)*HID;    // 512B aligned
  const unsigned short* bp = Bp + (size_t)lane*16;
  v8f c[8];
  #pragma unroll
  for (int nt = 0; nt < 8; ++nt) c[nt] = vzero();
  #pragma unroll 1
  for (int kt = 0; kt < 4; ++kt) {
    int kb = kt*32 + half*8;
    v4f A0 = *(const v4f*)(ar + kb),      A1 = *(const v4f*)(ar + kb + 4);
    v4f A2 = *(const v4f*)(ar + kb + 16), A3 = *(const v4f*)(ar + kb + 20);
    AFrag a;
    #pragma unroll
    for (int j = 0; j < 2; ++j) {
      a.u32[j]     = pk_bf(A0[2*j], A0[2*j+1]);
      a.u32[2 + j] = pk_bf(A1[2*j], A1[2*j+1]);
      a.u32[4 + j] = pk_bf(A2[2*j], A2[2*j+1]);
      a.u32[6 + j] = pk_bf(A3[2*j], A3[2*j+1]);
    }
    const unsigned short* bk = bp + (size_t)kt*4096;
    WMMA_GROUP4(bk, 0, a, c);
    WMMA_GROUP4(bk, 1, a, c);
  }
  #pragma unroll
  for (int i = 0; i < 8; ++i) {
    int n = tile*16 + half*8 + i;
    int g = batch[n];
    const float* xr = xWn + (size_t)n*HID;
    #pragma unroll
    for (int nt = 0; nt < 8; ++nt) {
      int col = nt*16 + m;
      float q = fmaxf(c[nt][i] + xr[col] + bn[col], 0.f);
      unsafeAtomicAdd(out + (size_t)g*HID + col, q);
    }
  }
}

extern "C" void kernel_launch(void* const* d_in, const int* in_sizes, int n_in,
                              void* d_out, int out_size, void* d_ws, size_t ws_size,
                              hipStream_t stream) {
  const float* x     = (const float*)d_in[0];
  const float* ea    = (const float*)d_in[1];
  const int*   eidx  = (const int*)  d_in[2];   // [2][NE]: row=eidx, col=eidx+NE
  const int*   batch = (const int*)  d_in[3];
  const float* We    = (const float*)d_in[4];
  const float* be    = (const float*)d_in[5];
  const float* Wc    = (const float*)d_in[6];
  const float* bc    = (const float*)d_in[7];
  const float* Wn    = (const float*)d_in[8];
  const float* bn    = (const float*)d_in[9];
  float* out = (float*)d_out;
  char*  ws  = (char*)d_ws;

  unsigned short* h   = (unsigned short*)(ws + OFF_H);
  unsigned short* h0  = (unsigned short*)(ws + OFF_H0);
  float*          agg = (float*)(ws + OFF_AGG);
  float*          xWe = (float*)(ws + OFF_XWE);
  float*          xWn = (float*)(ws + OFF_XWN);
  unsigned short* wp  = (unsigned short*)(ws + OFF_WP);

  const int* erow = eidx;
  const int* ecol = eidx + NE;

  k_pack<<<7, 256, 0, stream>>>(We, Wc, Wn, wp);
  k_zero4<<<(NG*HID/4 + 255)/256, 256, 0, stream>>>(out, NG*HID/4);

  // node-level precompute: xWe = x @ We[:133], xWn = x @ Wn[:133]
  k_node_gemm133<<<NN/(16*8), 256, 0, stream>>>(x, wp + WXP,  xWe);
  k_node_gemm133<<<NN/(16*8), 256, 0, stream>>>(x, wp + WNXP, xWn);

  // h0 = h = relu(xWe[row] + edge_attr @ We_e + be)
  k_edge_init<<<NE/(16*8), 256, 0, stream>>>(ea, erow, xWe, be, wp + WEEP, h, h0);

  for (int l = 0; l < 3; ++l) {
    k_zero4<<<(NN*HID/4 + 255)/256, 256, 0, stream>>>(agg, NN*HID/4);
    k_scatter<<<NE*16/256, 256, 0, stream>>>(h, ecol, agg);
    k_conv<<<NE/(16*8), 256, 0, stream>>>(agg, wp + WCP + l*16384, bc + l*HID,
                                          erow, h, h0);
  }

  // final: s = scatter(h), q = relu(xWn + s @ Wn_s + bn), out[batch] += q
  k_zero4<<<(NN*HID/4 + 255)/256, 256, 0, stream>>>(agg, NN*HID/4);
  k_scatter<<<NE*16/256, 256, 0, stream>>>(h, ecol, agg);
  k_final<<<NN/(16*8), 256, 0, stream>>>(agg, wp + WNSP, xWn, bn, batch, out);
}
